// Committee_64218351010273
// MI455X (gfx1250) — matrix operand order repaired and checked by reading, not compiled, not run on hardware
//
#include <hip/hip_runtime.h>

#define NUM_CLASSES 10
#define NMODELS     16
#define DDIM        512
#define NCOLS       160   // NMODELS * NUM_CLASSES
#define NTILES      10    // NCOLS / 16
#define KSTEPS      16    // DDIM / 32
#define WAVES_PER_BLOCK 4
#define LDS_STRIDE  164   // 164 % 64 = 36 -> 16 distinct banks across samples

typedef __attribute__((ext_vector_type(16))) __bf16 v16bf;
typedef __attribute__((ext_vector_type(8)))  float  v8f;
typedef __attribute__((ext_vector_type(4)))  float  f32x4;
typedef __attribute__((ext_vector_type(4)))  unsigned int u32x4;
typedef __attribute__((ext_vector_type(8)))  int    i32x8;
typedef __attribute__((ext_vector_type(4)))  int    i32x4;

// ---------------------------------------------------------------------------
// Prep: W [16][512][10] f32  ->  Wt [160][512] bf16  (j = m*10 + c, K contig)
// ---------------------------------------------------------------------------
__global__ void Committee_wprep(const float* __restrict__ W,
                                __bf16* __restrict__ Wt) {
    int idx = blockIdx.x * blockDim.x + threadIdx.x;
    if (idx >= NCOLS * DDIM) return;
    int j = idx / DDIM;
    int k = idx - j * DDIM;
    int m = j / NUM_CLASSES;
    int c = j - m * NUM_CLASSES;
    Wt[idx] = (__bf16)W[((size_t)m * DDIM + k) * NUM_CLASSES + c];
}

// ---------------------------------------------------------------------------
// TDM: DMA one K-slice of Wt (160 rows x 32 bf16, row stride 512) into LDS.
// D# per CDNA5 ISA ch.8: group0 = flags/lds_addr/global_addr/type,
// group1 = data_size/dims/tile dims/strides, groups 2-3 zero (2D tensor).
// ---------------------------------------------------------------------------
__device__ __forceinline__ void tdm_load_wslice(const __bf16* gsrc,
                                                unsigned lds_byte_addr) {
    unsigned long long ga = (unsigned long long)(uintptr_t)gsrc;
    u32x4 g0;
    g0[0] = 1u;                                         // count=1, user D#
    g0[1] = lds_byte_addr;                              // lds_addr
    g0[2] = (unsigned)(ga & 0xFFFFFFFFu);               // global_addr[31:0]
    g0[3] = (unsigned)((ga >> 32) & 0x01FFFFFFu)        // global_addr[56:32]
          | (2u << 30);                                 // type = 2 (image)
    i32x8 g1;
    g1[0] = 0x00010000;                                 // data_size = 2 bytes
    g1[1] = (int)(32u << 16);                           // tensor_dim0 = 32
    g1[2] = (int)(160u << 16);                          // tensor_dim1 = 160
    g1[3] = (int)(32u << 16);                           // tile_dim0 = 32
    g1[4] = 160;                                        // tile_dim1 = 160
    g1[5] = 512;                                        // dim0_stride = 512
    g1[6] = 0;
    g1[7] = 0;
    i32x4 z4 = {0, 0, 0, 0};
    i32x8 z8 = {0, 0, 0, 0, 0, 0, 0, 0};
    __builtin_amdgcn_tensor_load_to_lds(g0, g1, z4, z4, z8, 0);
}

// ---------------------------------------------------------------------------
// Main: per wave, a 16-sample tile x 160 columns via bf16 WMMA with W slices
// TDM-staged in LDS (double-buffered), then argmax + vote histogram.
// ---------------------------------------------------------------------------
__launch_bounds__(32 * WAVES_PER_BLOCK)
__global__ void Committee_main(const float* __restrict__ x,
                               const __bf16* __restrict__ Wt,
                               const float* __restrict__ bias,
                               float* __restrict__ out) {
    __shared__ __align__(16) __bf16 Wlds[2][NCOLS][32];     // 20 KB, dbl-buf
    __shared__ float lds[WAVES_PER_BLOCK][16][LDS_STRIDE];  // 42 KB logits

    const int wave = threadIdx.x >> 5;
    const int lane = threadIdx.x & 31;
    const int half = lane >> 4;      // 0: lanes 0-15, 1: lanes 16-31
    const int lrow = lane & 15;
    const int tile = blockIdx.x * WAVES_PER_BLOCK + wave;
    const int rowBase = tile * 16;

    const float* xrow = x + (size_t)(rowBase + lrow) * DDIM;

    v8f acc[NTILES];
#pragma unroll
    for (int t = 0; t < NTILES; ++t) acc[t] = (v8f)0.0f;

    // Prefetch K-slice 0 via the Tensor Data Mover (wave 0 only; EXEC ignored).
    if (wave == 0) {
        tdm_load_wslice(Wt, (unsigned)(uintptr_t)&Wlds[0][0][0]);
    }

    for (int ks = 0; ks < KSTEPS; ++ks) {
        const int buf = ks & 1;
        // Wait for the TDM slice feeding this iteration, then publish it.
        if (wave == 0) __builtin_amdgcn_s_wait_tensorcnt(0);
        __syncthreads();   // slice[buf] visible; slice[buf^1] no longer read
        // Overlap: DMA next slice while this iteration computes.
        if (wave == 0 && ks + 1 < KSTEPS) {
            tdm_load_wslice(Wt + (ks + 1) * 32,
                            (unsigned)(uintptr_t)&Wlds[buf ^ 1][0][0]);
        }

        const int k0 = ks * 32;
        // A fragment (16x32 bf16): lanes 0-15 hold K k0+0..15,
        // lanes 16-31 hold K k0+16..31 of row (rowBase+lrow).
        const int ka = k0 + half * 16;
        const int kb = k0 + half * 16 + 8;
        f32x4 f0 = *(const f32x4*)(xrow + ka);
        f32x4 f1 = *(const f32x4*)(xrow + ka + 4);
        f32x4 f2 = *(const f32x4*)(xrow + kb);
        f32x4 f3 = *(const f32x4*)(xrow + kb + 4);
        v16bf a;
#pragma unroll
        for (int i = 0; i < 4; ++i) {
            a[i]      = (__bf16)f0[i];
            a[4 + i]  = (__bf16)f1[i];
            a[8 + i]  = (__bf16)f2[i];
            a[12 + i] = (__bf16)f3[i];
        }
        // B fragments (32x16 bf16) from LDS: lane n = column t*16+lrow,
        // lanes 0-15 hold K k0..k0+15, lanes 16-31 hold K k0+16..k0+31.
        // One-deep software pipeline: keep two ds_load_b128 pairs in flight
        // so WMMA issue waits on DScnt<=2 instead of DScnt==0.
        const __bf16* wl = &Wlds[buf][lrow][half * 16];
        v16bf bf_cur = *(const v16bf*)(wl);
#pragma unroll
        for (int t = 0; t < NTILES; ++t) {
            v16bf bf_next;
            if (t + 1 < NTILES)
                bf_next = *(const v16bf*)(wl + (t + 1) * 16 * 32);
            acc[t] = __builtin_amdgcn_wmma_f32_16x16x32_bf16(
                /*neg_a=*/false, a, /*neg_b=*/false, bf_cur,
                /*c_mod=*/(short)0, acc[t],
                /*reuse_a=*/false, /*reuse_b=*/false);
            bf_cur = bf_next;
        }
    }

    // Scatter C tiles to LDS: VGPR v holds sample (v + half*8), lane holds
    // column j = t*16 + lrow.
#pragma unroll
    for (int t = 0; t < NTILES; ++t) {
        const int j = t * 16 + lrow;
#pragma unroll
        for (int v = 0; v < 8; ++v) {
            lds[wave][v + half * 8][j] = acc[t][v];
        }
    }
    __syncthreads();

    // Lane s (and s+16) handle models [half*8, half*8+8) of sample s.
    float cnt[NUM_CLASSES];
#pragma unroll
    for (int c = 0; c < NUM_CLASSES; ++c) cnt[c] = 0.0f;

    const float* srow = &lds[wave][lrow][0];
    for (int mm = 0; mm < 8; ++mm) {
        const int m = half * 8 + mm;
        float best = -3.4e38f;
        int bi = 0;
#pragma unroll
        for (int c = 0; c < NUM_CLASSES; ++c) {
            const int j = m * NUM_CLASSES + c;
            float v = srow[j] + bias[j];
            if (v > best) { best = v; bi = c; }
        }
#pragma unroll
        for (int c = 0; c < NUM_CLASSES; ++c)
            cnt[c] += (bi == c) ? 1.0f : 0.0f;
    }

    // Combine the two half-wave partial histograms (wave32 xor-shuffle).
#pragma unroll
    for (int c = 0; c < NUM_CLASSES; ++c)
        cnt[c] += __shfl_xor(cnt[c], 16, 32);

    if (half == 0) {
        float* orow = out + (size_t)(rowBase + lrow) * NUM_CLASSES;
#pragma unroll
        for (int c = 0; c < NUM_CLASSES; ++c) orow[c] = cnt[c];
    }
}

// ---------------------------------------------------------------------------
extern "C" void kernel_launch(void* const* d_in, const int* in_sizes, int n_in,
                              void* d_out, int out_size, void* d_ws, size_t ws_size,
                              hipStream_t stream) {
    const float* x = (const float*)d_in[0];   // [B, 512]
    const float* W = (const float*)d_in[1];   // [16, 512, 10]
    const float* b = (const float*)d_in[2];   // [16, 10]
    float* out = (float*)d_out;               // [B, 10]
    __bf16* Wt = (__bf16*)d_ws;               // [160][512] bf16 = 160 KB

    const int prepN = NCOLS * DDIM;
    Committee_wprep<<<(prepN + 255) / 256, 256, 0, stream>>>(W, Wt);

    const int B = in_sizes[0] / DDIM;         // 65536
    const int tiles = B / 16;                 // 4096
    Committee_main<<<tiles / WAVES_PER_BLOCK, 32 * WAVES_PER_BLOCK, 0, stream>>>(
        x, Wt, b, out);
}
